// TensoSDFShapeRenderer_3143916061279
// MI455X (gfx1250) — matrix operand order repaired and checked by reading, not compiled, not run on hardware
//
#include <hip/hip_runtime.h>
#include <hip/hip_bf16.h>
#include <cstdint>

typedef __attribute__((ext_vector_type(16))) _Float16 v16h;
typedef __attribute__((ext_vector_type(8)))  _Float16 v8h;
typedef __attribute__((ext_vector_type(8)))  float    v8f;

#define N_COMP_   36
#define GRID_R    300
#define SDF_DIM_  256
#define MULTIRES_ 3
#define N_PTS_    524288
#define EMB_CH_   21
#define IN_CH_    129
#define OUT_CH_   129
#define K1P       160      // IN_CH padded to 5 k-steps of 32
#define K2        256      // SDF_DIM: 8 k-steps of 32
#define W2_ROWS   144      // OUT_CH padded to 9 n-tiles of 16
#define M_TILE    128
#define H_STRIDE  264      // SDF_DIM + 8 halfs pad (bank-conflict dodge)

// ---------------------------------------------------------------------------
// Kernel 0: convert f32 weights to f16, pre-padded into WMMA-friendly layouts.
//   W1h[256][160]: W1h[n][k] = w1[n][k] (k<129) else 0   (B for GEMM1: B[k][n]=w1[n][k])
//   W2h[144][256]: W2h[n][k] = w2[n][k] (n<129) else 0   (B for GEMM2)
// ---------------------------------------------------------------------------
__global__ void prep_weights(const float* __restrict__ w1, const float* __restrict__ w2,
                             _Float16* __restrict__ W1h, _Float16* __restrict__ W2h) {
  int i = blockIdx.x * blockDim.x + threadIdx.x;
  if (i < SDF_DIM_ * K1P) {
    int n = i / K1P, k = i - n * K1P;
    W1h[i] = (k < IN_CH_) ? (_Float16)w1[n * IN_CH_ + k] : (_Float16)0.0f;
  } else {
    int j = i - SDF_DIM_ * K1P;
    if (j < W2_ROWS * K2) {
      int n = j / K2, k = j - n * K2;
      W2h[j] = (n < OUT_CH_) ? (_Float16)w2[n * K2 + k] : (_Float16)0.0f;
    }
  }
}

// Branchless, numerically stable softplus: max(z,0) + log(1 + exp(-|z|)).
// Lowers to straight-line v_max/v_exp/v_log (TRANS ops co-execute with WMMA).
__device__ __forceinline__ float softplus_fast(float z) {
  return fmaxf(z, 0.0f) + __logf(1.0f + __expf(-fabsf(z)));
}

// bilinear plane tap * linear line tap for 36 components
__device__ __forceinline__ void sample_planeline(
    const float* __restrict__ pl, const float* __restrict__ ln,
    int u0, int v0, float tu, float tv, int w0, float tw,
    _Float16* __restrict__ dst) {
  const float w00 = (1.f - tu) * (1.f - tv);
  const float w01 = tu * (1.f - tv);
  const float w10 = (1.f - tu) * tv;
  const float w11 = tu * tv;
  #pragma unroll 4
  for (int c = 0; c < N_COMP_; ++c) {
    const float* p = pl + c * (GRID_R * GRID_R) + v0 * GRID_R + u0;
    float pf = p[0] * w00 + p[1] * w01 + p[GRID_R] * w10 + p[GRID_R + 1] * w11;
    const float* l = ln + c * GRID_R + w0;
    float lf = l[0] * (1.f - tw) + l[1] * tw;
    dst[c] = (_Float16)(pf * lf);
  }
}

// ---------------------------------------------------------------------------
// Fused kernel: gather + embed -> GEMM1(WMMA f16) -> softplus -> GEMM2(WMMA f16)
// ---------------------------------------------------------------------------
__global__ __launch_bounds__(256) void tensosdf_fused(
    const float* __restrict__ xyz,
    const float* __restrict__ planes,
    const float* __restrict__ lines,
    const _Float16* __restrict__ W1h,
    const float* __restrict__ b1,
    const _Float16* __restrict__ W2h,
    const float* __restrict__ b2,
    float* __restrict__ out) {

  __shared__ _Float16 sIn[M_TILE * K1P];      // 40960 B
  __shared__ _Float16 sH [M_TILE * H_STRIDE]; // 67584 B

  const int tid  = threadIdx.x;
  const int pt   = tid & (M_TILE - 1);
  const int part = tid >> 7;                  // 0 or 1 (two threads per point)
  const int gpt  = blockIdx.x * M_TILE + pt;

  // ---- Stage 1: sampling + positional encoding into sIn -------------------
  const float x = xyz[gpt * 3 + 0];
  const float y = xyz[gpt * 3 + 1];
  const float z = xyz[gpt * 3 + 2];

  const float gs = 0.5f * (float)(GRID_R - 1);
  float fx = (x + 1.f) * gs; int ix = min(max((int)floorf(fx), 0), GRID_R - 2); float tx = fx - (float)ix;
  float fy = (y + 1.f) * gs; int iy = min(max((int)floorf(fy), 0), GRID_R - 2); float ty = fy - (float)iy;
  float fz = (z + 1.f) * gs; int iz = min(max((int)floorf(fz), 0), GRID_R - 2); float tz = fz - (float)iz;

  _Float16* row = sIn + pt * K1P;
  if (part == 0) {
    // positional encoding: [x,y,z, sin(c*2^f) coord-major, cos(...)]
    row[0] = (_Float16)x; row[1] = (_Float16)y; row[2] = (_Float16)z;
    const float c3[3] = {x, y, z};
    #pragma unroll
    for (int ci = 0; ci < 3; ++ci) {
      #pragma unroll
      for (int f = 0; f < MULTIRES_; ++f) {
        float a = c3[ci] * (float)(1 << f);
        row[3 + ci * MULTIRES_ + f]     = (_Float16)__sinf(a);
        row[3 + 9 + ci * MULTIRES_ + f] = (_Float16)__cosf(a);
      }
    }
    #pragma unroll
    for (int k = IN_CH_; k < K1P; ++k) row[k] = (_Float16)0.0f;  // zero K-pad
    // plane 0: (x,y) grid, line z
    sample_planeline(planes, lines, ix, iy, tx, ty, iz, tz, row + EMB_CH_);
  } else {
    // plane 1: (x,z) grid, line y
    sample_planeline(planes + 1 * N_COMP_ * GRID_R * GRID_R, lines + 1 * N_COMP_ * GRID_R,
                     ix, iz, tx, tz, iy, ty, row + EMB_CH_ + N_COMP_);
    // plane 2: (y,z) grid, line x
    sample_planeline(planes + 2 * N_COMP_ * GRID_R * GRID_R, lines + 2 * N_COMP_ * GRID_R,
                     iy, iz, ty, tz, ix, tx, row + EMB_CH_ + 2 * N_COMP_);
  }
  __syncthreads();   // only barrier: afterwards each wave owns its 16 rows

  // ---- Stage 2: GEMM1 h = softplus(100*(in @ w1^T + b1))/100 --------------
  const int lane = tid & 31;
  const int wv   = tid >> 5;       // 8 waves -> 8 M-tiles of 16 rows
  const int m0   = wv * 16;
  const int hf   = lane >> 4;      // wave half
  const int mr   = lane & 15;

  // A fragments (16x32 f16, contiguous k order chosen here, same as B): lanes 0-15 hold K {0..15}, lanes 16-31 hold K {16..31}
  v16h a1[5];
  #pragma unroll
  for (int ks = 0; ks < 5; ++ks) {
    const _Float16* p = sIn + (m0 + mr) * K1P + ks * 32 + hf * 16;
    v8h lo = *(const v8h*)p;
    v8h hi = *(const v8h*)(p + 8);
    v16h a;
    #pragma unroll
    for (int e = 0; e < 8; ++e) { a[e] = lo[e]; a[e + 8] = hi[e]; }
    a1[ks] = a;
  }

  #pragma unroll
  for (int nt = 0; nt < 16; ++nt) {
    const int n = nt * 16 + mr;               // output channel of this lane
    v8f c = {};
    #pragma unroll
    for (int ks = 0; ks < 5; ++ks) {
      // B fragment (32x16): lane holds column n, K = k0 + hf*16 + {0..15}, contiguous in W1h
      const _Float16* p = W1h + n * K1P + ks * 32 + hf * 16;
      v16h b = *(const v16h*)p;
      c = __builtin_amdgcn_wmma_f32_16x16x32_f16(false, a1[ks], false, b,
                                                 (short)0, c, false, false);
    }
    const float bias = b1[n];
    #pragma unroll
    for (int v = 0; v < 8; ++v) {
      // C/D layout: VGPR v -> row m0 + v + 8*hf, col n
      float sp = softplus_fast(100.f * (c[v] + bias));
      sH[(m0 + v + hf * 8) * H_STRIDE + n] = (_Float16)(sp * 0.01f);
    }
  }

  // ---- Stage 3: GEMM2 out = h @ w2^T + b2 ---------------------------------
  v16h a2[8];
  #pragma unroll
  for (int ks = 0; ks < 8; ++ks) {
    const _Float16* p = sH + (m0 + mr) * H_STRIDE + ks * 32 + hf * 16;
    v8h lo = *(const v8h*)p;
    v8h hi = *(const v8h*)(p + 8);
    v16h a;
    #pragma unroll
    for (int e = 0; e < 8; ++e) { a[e] = lo[e]; a[e + 8] = hi[e]; }
    a2[ks] = a;
  }

  #pragma unroll
  for (int nt = 0; nt < 9; ++nt) {
    const int n = nt * 16 + mr;
    v8f c = {};
    #pragma unroll
    for (int ks = 0; ks < 8; ++ks) {
      const _Float16* p = W2h + n * K2 + ks * 32 + hf * 16;
      v16h b = *(const v16h*)p;
      c = __builtin_amdgcn_wmma_f32_16x16x32_f16(false, a2[ks], false, b,
                                                 (short)0, c, false, false);
    }
    if (n < OUT_CH_) {
      const float bias = b2[n];
      #pragma unroll
      for (int v = 0; v < 8; ++v) {
        size_t r = (size_t)(blockIdx.x * M_TILE + m0 + v + hf * 8);
        out[r * OUT_CH_ + n] = c[v] + bias;   // lanes 0-15 -> 16 contiguous cols
      }
    }
  }
}

// ---------------------------------------------------------------------------
extern "C" void kernel_launch(void* const* d_in, const int* in_sizes, int n_in,
                              void* d_out, int out_size, void* d_ws, size_t ws_size,
                              hipStream_t stream) {
  const float* xyz    = (const float*)d_in[0];
  const float* planes = (const float*)d_in[1];
  const float* lines  = (const float*)d_in[2];
  const float* w1     = (const float*)d_in[3];
  const float* b1     = (const float*)d_in[4];
  const float* w2     = (const float*)d_in[5];
  const float* b2     = (const float*)d_in[6];
  float* out = (float*)d_out;

  _Float16* W1h = (_Float16*)d_ws;                                       // 81920 B
  _Float16* W2h = (_Float16*)((char*)d_ws + (size_t)SDF_DIM_ * K1P * 2); // 73728 B

  const int prep_total = SDF_DIM_ * K1P + W2_ROWS * K2;
  prep_weights<<<(prep_total + 255) / 256, 256, 0, stream>>>(w1, w2, W1h, W2h);

  tensosdf_fused<<<N_PTS_ / M_TILE, 256, 0, stream>>>(
      xyz, planes, lines, W1h, b1, W2h, b2, out);
}
